// GCNEncoder_52716428591567
// MI455X (gfx1250) — compile-verified, matrix operand order checked by reading, not run on hardware
//
#include <hip/hip_runtime.h>

// GCN encoder: N nodes, E edges, 128 -> 64 -> {32, 32}
#define F_IN  128
#define F_HID 64
#define F_OUT 32

typedef __attribute__((ext_vector_type(16))) _Float16 v16h;
typedef __attribute__((ext_vector_type(8)))  float    v8f;

// K-index mapping for 16-bit A/B fragments of V_WMMA_*_16X16X32_F16
// (ISA 7.12.2: VGPR v holds K = 2v,2v+1 for lanes 0-15, +8 for lanes 16-31;
//  VGPRs 4-7 cover K=16..31 the same way).
__device__ __forceinline__ int kmap16(int e, int half) {
    int v = e >> 1, pos = e & 1;
    return ((v < 4) ? 0 : 16) + half * 8 + ((v & 3) << 1) + pos;
}

// Pre-pack W[K,NC] (f32 row-major) into per-fragment order:
// Wp[((kt*NT + nt)*32 + lane)*16 + e] = f16(W[k, nt*16 + lane%16]), k = kt*32 + kmap16(e, lane/16)
// so the GEMM loads each lane's B fragment as one contiguous 32B v16h.
template <int K, int NC>
__global__ void k_pack_w(const float* __restrict__ W, _Float16* __restrict__ Wp) {
    constexpr int NT = NC / 16;
    int idx = blockIdx.x * blockDim.x + threadIdx.x;
    if (idx >= K * NC) return;
    int e    = idx & 15;
    int lane = (idx >> 4) & 31;
    int t    = idx >> 9;              // kt*NT + nt
    int nt   = t % NT, kt = t / NT;
    int half = lane >> 4;
    int n    = nt * 16 + (lane & 15);
    int k    = kt * 32 + kmap16(e, half);
    Wp[idx]  = (_Float16)W[k * NC + n];
}

// C = A[M,K] (f32, row-major) x W (fragment-packed f16), f16 WMMA, f32 accum.
// One wave -> one 16-row M tile x all NC columns. Block = 256 threads = 8 waves.
template <int K, int NC>
__global__ __launch_bounds__(256) void gemm_f16_wmma(const float* __restrict__ A,
                                                     const _Float16* __restrict__ Wp,
                                                     float* __restrict__ C, int M) {
    constexpr int NT = NC / 16;
    const int wave  = threadIdx.x >> 5;
    const int lane  = threadIdx.x & 31;
    const int mtile = blockIdx.x * 8 + wave;
    const int mbase = mtile * 16;
    if (mbase >= M) return;                 // wave-uniform: EXEC stays all-ones
    const int half = lane >> 4;
    const int l16  = lane & 15;

    const v16h* __restrict__ WpV = (const v16h*)Wp;
    const float* __restrict__ arow = A + (size_t)(mbase + l16) * K;

    v8f acc[NT] = {};

#pragma unroll
    for (int kt = 0; kt < K / 32; ++kt) {
        // A fragment: two contiguous 8-float runs -> global_load_b128 pairs.
        v16h a;
#pragma unroll
        for (int e = 0; e < 8; ++e) {
            a[e]     = (_Float16)arow[kt * 32 + half * 8 + e];
            a[e + 8] = (_Float16)arow[kt * 32 + 16 + half * 8 + e];
        }
#pragma unroll
        for (int nt = 0; nt < NT; ++nt) {
            v16h b = WpV[(kt * NT + nt) * 32 + lane];   // contiguous 32B per lane
            acc[nt] = __builtin_amdgcn_wmma_f32_16x16x32_f16(
                false, a, false, b, (short)0, acc[nt], false, false);
        }
    }

    // D layout: VGPR j -> row (half*8 + j), col = lane%16 within the N tile.
#pragma unroll
    for (int nt = 0; nt < NT; ++nt) {
#pragma unroll
        for (int j = 0; j < 8; ++j) {
            int m = mbase + half * 8 + j;
            C[(size_t)m * NC + nt * 16 + l16] = acc[nt][j];
        }
    }
}

// ---- graph / elementwise kernels -------------------------------------------

__global__ void k_init_deg(float* deg, int n) {
    int i = blockIdx.x * blockDim.x + threadIdx.x;
    if (i < n) deg[i] = 1.0f;               // +1 self-loop
}

__global__ void k_edge_deg(const long long* __restrict__ ei, float* deg, int E) {
    int e = blockIdx.x * blockDim.x + threadIdx.x;
    if (e < E) atomicAdd(&deg[(int)ei[(size_t)E + e]], 1.0f);
}

__global__ void k_rsqrt_inplace(float* d, int n) {
    int i = blockIdx.x * blockDim.x + threadIdx.x;
    if (i < n) d[i] = rsqrtf(d[i]);
}

// out[i, 4f..4f+3] = pre * dinv[i]^2 + bias     (fully overwrites out)
__global__ void k_selfloop_bias4(const float4* __restrict__ pre,
                                 const float* __restrict__ dinv,
                                 const float4* __restrict__ bias,
                                 float4* __restrict__ out, int n, int F4) {
    long long idx = (long long)blockIdx.x * blockDim.x + threadIdx.x;
    if (idx >= (long long)n * F4) return;
    int i = (int)(idx / F4), f = (int)(idx - (long long)i * F4);
    float di = dinv[i], s = di * di;
    float4 p = pre[idx], b = bias[f];
    out[idx] = make_float4(p.x * s + b.x, p.y * s + b.y, p.z * s + b.z, p.w * s + b.w);
}

// out[dst, 4f..4f+3] += pre[src, 4f..4f+3] * dinv[src]*dinv[dst]
__global__ void k_edge_agg4(const long long* __restrict__ ei,
                            const float* __restrict__ dinv,
                            const float* __restrict__ pre,
                            float* __restrict__ out, int E, int F4, int F) {
    long long idx = (long long)blockIdx.x * blockDim.x + threadIdx.x;
    if (idx >= (long long)E * F4) return;
    int e = (int)(idx / F4), fq = (int)(idx - (long long)e * F4);
    int s = (int)ei[e];
    int d = (int)ei[(size_t)E + e];
    float c = dinv[s] * dinv[d];
    float4 p = ((const float4*)(pre + (size_t)s * F))[fq];
    float* o = out + (size_t)d * F + fq * 4;
    atomicAdd(o + 0, p.x * c);
    atomicAdd(o + 1, p.y * c);
    atomicAdd(o + 2, p.z * c);
    atomicAdd(o + 3, p.w * c);
}

// fused dual aggregation for mu / logstd (share edge metadata and coef)
__global__ void k_edge_agg2x4(const long long* __restrict__ ei,
                              const float* __restrict__ dinv,
                              const float* __restrict__ pmu,
                              const float* __restrict__ pls,
                              float* __restrict__ omu,
                              float* __restrict__ ols, int E, int F4, int F) {
    long long idx = (long long)blockIdx.x * blockDim.x + threadIdx.x;
    if (idx >= (long long)E * F4) return;
    int e = (int)(idx / F4), fq = (int)(idx - (long long)e * F4);
    int s = (int)ei[e];
    int d = (int)ei[(size_t)E + e];
    float c = dinv[s] * dinv[d];
    float4 pm = ((const float4*)(pmu + (size_t)s * F))[fq];
    float4 pl = ((const float4*)(pls + (size_t)s * F))[fq];
    float* om = omu + (size_t)d * F + fq * 4;
    float* ol = ols + (size_t)d * F + fq * 4;
    atomicAdd(om + 0, pm.x * c);
    atomicAdd(om + 1, pm.y * c);
    atomicAdd(om + 2, pm.z * c);
    atomicAdd(om + 3, pm.w * c);
    atomicAdd(ol + 0, pl.x * c);
    atomicAdd(ol + 1, pl.y * c);
    atomicAdd(ol + 2, pl.z * c);
    atomicAdd(ol + 3, pl.w * c);
}

__global__ void k_relu4(float4* x, long long n4) {
    long long i = (long long)blockIdx.x * blockDim.x + threadIdx.x;
    if (i < n4) {
        float4 v = x[i];
        x[i] = make_float4(fmaxf(v.x, 0.f), fmaxf(v.y, 0.f), fmaxf(v.z, 0.f), fmaxf(v.w, 0.f));
    }
}

// ---- driver -----------------------------------------------------------------

extern "C" void kernel_launch(void* const* d_in, const int* in_sizes, int n_in,
                              void* d_out, int out_size, void* d_ws, size_t ws_size,
                              hipStream_t stream) {
    const float*     x    = (const float*)d_in[0];
    const long long* ei   = (const long long*)d_in[1];   // int64 [2, E]
    const float*     W1   = (const float*)d_in[2];
    const float*     b1   = (const float*)d_in[3];
    const float*     Wmu  = (const float*)d_in[4];
    const float*     bmu  = (const float*)d_in[5];
    const float*     Wls  = (const float*)d_in[6];
    const float*     bls  = (const float*)d_in[7];
    float*           out  = (float*)d_out;

    const int N = in_sizes[0] / F_IN;
    const int E = in_sizes[1] / 2;

    // workspace layout (floats)
    float* ws   = (float*)d_ws;
    float* dinv = ws;                               // N
    float* hw   = dinv + N;                         // N * F_HID  (x @ W1)
    float* h    = hw + (size_t)N * F_HID;           // N * F_HID  (aggregated, relu)
    float* mup  = h + (size_t)N * F_HID;            // N * F_OUT  (h @ Wmu)
    float* lsp  = mup + (size_t)N * F_OUT;          // N * F_OUT  (h @ Wls)

    // fragment-packed f16 weights, 32B-aligned
    size_t woff = (size_t)N * (1 + F_HID + F_HID + F_OUT + F_OUT);
    woff = (woff + 7) & ~(size_t)7;
    _Float16* w1p  = (_Float16*)(ws + woff);        // F_IN * F_HID
    _Float16* wmup = w1p + F_IN * F_HID;            // F_HID * F_OUT
    _Float16* wlsp = wmup + F_HID * F_OUT;          // F_HID * F_OUT

    float* out_mu = out;
    float* out_ls = out + (size_t)N * F_OUT;

    const int TB = 256;
    auto blocks = [&](long long total) { return (int)((total + TB - 1) / TB); };

    // 0. pack weights into WMMA fragment order (tiny)
    k_pack_w<F_IN, F_HID><<<blocks(F_IN * F_HID), TB, 0, stream>>>(W1, w1p);
    k_pack_w<F_HID, F_OUT><<<blocks(F_HID * F_OUT), TB, 0, stream>>>(Wmu, wmup);
    k_pack_w<F_HID, F_OUT><<<blocks(F_HID * F_OUT), TB, 0, stream>>>(Wls, wlsp);

    // 1. degree (with self loop) -> dinv_sqrt
    k_init_deg<<<blocks(N), TB, 0, stream>>>(dinv, N);
    k_edge_deg<<<blocks(E), TB, 0, stream>>>(ei, dinv, E);
    k_rsqrt_inplace<<<blocks(N), TB, 0, stream>>>(dinv, N);

    // 2. hw = x @ W1   (WMMA)
    {
        int mtiles = (N + 15) / 16;
        int grid = (mtiles + 7) / 8;
        gemm_f16_wmma<F_IN, F_HID><<<grid, TB, 0, stream>>>(x, w1p, hw, N);
    }

    // 3. h = scatter-add(hw, edges) + hw*dinv^2 + b1 ; relu
    k_selfloop_bias4<<<blocks((long long)N * (F_HID / 4)), TB, 0, stream>>>(
        (const float4*)hw, dinv, (const float4*)b1, (float4*)h, N, F_HID / 4);
    k_edge_agg4<<<blocks((long long)E * (F_HID / 4)), TB, 0, stream>>>(
        ei, dinv, hw, h, E, F_HID / 4, F_HID);
    k_relu4<<<blocks((long long)N * (F_HID / 4)), TB, 0, stream>>>(
        (float4*)h, (long long)N * (F_HID / 4));

    // 4. mup = h @ Wmu ; lsp = h @ Wls   (WMMA)
    {
        int mtiles = (N + 15) / 16;
        int grid = (mtiles + 7) / 8;
        gemm_f16_wmma<F_HID, F_OUT><<<grid, TB, 0, stream>>>(h, wmup, mup, N);
        gemm_f16_wmma<F_HID, F_OUT><<<grid, TB, 0, stream>>>(h, wlsp, lsp, N);
    }

    // 5. mu / logstd: self-loop + bias init, then fused edge scatter-add
    k_selfloop_bias4<<<blocks((long long)N * (F_OUT / 4)), TB, 0, stream>>>(
        (const float4*)mup, dinv, (const float4*)bmu, (float4*)out_mu, N, F_OUT / 4);
    k_selfloop_bias4<<<blocks((long long)N * (F_OUT / 4)), TB, 0, stream>>>(
        (const float4*)lsp, dinv, (const float4*)bls, (float4*)out_ls, N, F_OUT / 4);
    k_edge_agg2x4<<<blocks((long long)E * (F_OUT / 4)), TB, 0, stream>>>(
        ei, dinv, mup, lsp, out_mu, out_ls, E, F_OUT / 4, F_OUT);
}
